// AddingToQ_67405216743425
// MI455X (gfx1250) — compile-verified, matrix operand order checked by reading, not run on hardware
//
#include <hip/hip_runtime.h>
#include <hip/hip_fp16.h>

typedef __attribute__((ext_vector_type(16))) _Float16 v16h;
typedef __attribute__((ext_vector_type(8)))  float    v8f;

#define B_   256
#define NQ_  15
#define NC_  30
#define NPG_ 60
#define NN_  (B_ * NPG_)          // 15360 nodes
#define EQ_  45
#define EC_  90
#define EE_  (B_ * (EQ_ + EC_ + NQ_ * (NC_ - NQ_)))  // 92160 edges
#define D_   128
#define T_   64
#define NPROP_ 5
#define SKIT_  10

// ---------------------------------------------------------------------------
// WMMA fragment helpers (CDNA5 16-bit A/B layout, wave32).
// element i of a lane's fragment -> K = (i&7) + ((i>>3)<<4) + (half<<3)
// (two contiguous 8-half runs -> ds_load_b128 pairs from LDS)
// ---------------------------------------------------------------------------
__device__ __forceinline__ int frag_k(int i, int half) {
  return (i & 7) + ((i >> 3) << 4) + (half << 3);
}

__device__ __forceinline__ v16h frag_a_lds(const _Float16* sm, int ld, int row0, int kb) {
  int lane = threadIdx.x & 31;
  int row  = row0 + (lane & 15);
  int half = lane >> 4;
  v16h a;
#pragma unroll
  for (int i = 0; i < 16; ++i) a[i] = sm[row * ld + kb + frag_k(i, half)];
  return a;
}

// B from LDS stored N x K row-major (i.e. B^T): B[k][n] = sm[n*ld + k]
__device__ __forceinline__ v16h frag_bt_lds(const _Float16* sm, int ld, int kb, int cb) {
  int lane = threadIdx.x & 31;
  int col  = cb + (lane & 15);
  int half = lane >> 4;
  v16h b;
#pragma unroll
  for (int i = 0; i < 16; ++i) b[i] = sm[col * ld + kb + frag_k(i, half)];
  return b;
}

// B from global pre-swizzled fragment-ready layout: each lane's 16 halfs are
// contiguous (32B aligned) -> one v16h load = 2x global_load_b128.
__device__ __forceinline__ v16h frag_b_swz(const _Float16* wsw, int ntc, int kb, int cb) {
  const v16h* p = (const v16h*)wsw;
  return p[((size_t)((kb >> 5) * ntc + (cb >> 4))) * 32 + (threadIdx.x & 31)];
}

#define WMMA_F16(a, b, c) \
  __builtin_amdgcn_wmma_f32_16x16x32_f16(false, (a), false, (b), (short)0, (c), false, false)

// ---------------------------------------------------------------------------
// Weight swizzle: W (K x ncols, f32 row-major) -> fragment-ready f16.
// out[((kt*ntc + ct)*32 + lane)*16 + elem] = W[kt*32 + frag_k(elem, lane>>4)][ct*16 + (lane&15)]
// ---------------------------------------------------------------------------
__global__ void swizzle_w_kernel(const float* __restrict__ w, _Float16* __restrict__ out,
                                 int K, int ncols) {
  int i = blockIdx.x * blockDim.x + threadIdx.x;
  if (i >= K * ncols) return;
  int elem = i & 15;
  int lane = (i >> 4) & 31;
  int tile = i >> 9;
  int ntc  = ncols >> 4;
  int kt = tile / ntc, ct = tile - kt * ntc;
  int half = lane >> 4;
  int k   = kt * 32 + frag_k(elem, half);
  int col = ct * 16 + (lane & 15);
  out[i] = (_Float16)w[k * ncols + col];
}

__global__ void zero_kernel(float* __restrict__ p, int n) {
  int i = blockIdx.x * blockDim.x + threadIdx.x;
  if (i < n) p[i] = 0.0f;
}

// Node encoder: h (f32) + h16 (f16).  x has feature dim 1.
__global__ void encode_node_kernel(const float* __restrict__ x, const float* __restrict__ w,
                                   const float* __restrict__ bias, float* __restrict__ h,
                                   _Float16* __restrict__ h16, int n) {
  int i = blockIdx.x * blockDim.x + threadIdx.x;
  if (i >= n * D_) return;
  int row = i >> 7, col = i & (D_ - 1);
  float v = x[row] * w[col] + bias[col];
  h[i] = v;
  h16[i] = (_Float16)v;
}

// Edge encoder: e16 only (f16 is the only consumer).
__global__ void encode_edge_kernel(const float* __restrict__ x, const float* __restrict__ w,
                                   const float* __restrict__ bias, _Float16* __restrict__ e16,
                                   int n) {
  int i = blockIdx.x * blockDim.x + threadIdx.x;
  if (i >= n * D_) return;
  int row = i >> 7, col = i & (D_ - 1);
  e16[i] = (_Float16)(x[row] * w[col] + bias[col]);
}

// ---------------------------------------------------------------------------
// Message MLP (fused) + mask + scatter-add into agg.
// Block = 16 edges, 256 threads (8 waves).
// ---------------------------------------------------------------------------
__global__ __launch_bounds__(256) void msg_kernel(
    const _Float16* __restrict__ h16, const _Float16* __restrict__ e16,
    const int* __restrict__ from_idx, const int* __restrict__ to_idx,
    const float* __restrict__ mask,
    const _Float16* __restrict__ w1, const float* __restrict__ b1,
    const _Float16* __restrict__ w2, const float* __restrict__ b2,
    float* __restrict__ agg) {
  __shared__ __align__(16) _Float16 sA[16 * 384];
  __shared__ __align__(16) _Float16 sH[16 * 256];
  __shared__ int   sFrom[16];
  __shared__ int   sTo[16];
  __shared__ float sM[16];

  int tid = threadIdx.x;
  int e0  = blockIdx.x * 16;

  if (tid < 16) {
    sFrom[tid] = from_idx[e0 + tid];
    sTo[tid]   = to_idx[e0 + tid];
    sM[tid]    = mask[e0 + tid];
  }
  if (tid == 0 && e0 + 16 < EE_)
    __builtin_prefetch(&e16[(size_t)(e0 + 16) * D_], 0, 0);
  __syncthreads();

  // Stage A = [h16[from] | h16[to] | e16] as 16B chunks (48 uint4 per row).
  {
    uint4* dst = (uint4*)sA;
    for (int t = tid; t < 16 * 48; t += 256) {
      int row = t / 48, part = t - row * 48;
      uint4 v;
      if (part < 16)      v = ((const uint4*)(h16 + (size_t)sFrom[row] * D_))[part];
      else if (part < 32) v = ((const uint4*)(h16 + (size_t)sTo[row] * D_))[part - 16];
      else                v = ((const uint4*)(e16 + (size_t)(e0 + row) * D_))[part - 32];
      dst[t] = v;
    }
  }
  __syncthreads();

  int wave = tid >> 5, lane = tid & 31, half = lane >> 4;

  // ---- GEMM1: 16x384 @ 384x256, each wave owns 32 hidden columns
  v8f c0 = {}, c1 = {};
  int cb0 = wave * 32;
#pragma unroll
  for (int kb = 0; kb < 384; kb += 32) {
    v16h a  = frag_a_lds(sA, 384, 0, kb);
    v16h b0 = frag_b_swz(w1, 16, kb, cb0);
    v16h b1v = frag_b_swz(w1, 16, kb, cb0 + 16);
    c0 = WMMA_F16(a, b0, c0);
    c1 = WMMA_F16(a, b1v, c1);
  }
#pragma unroll
  for (int r = 0; r < 8; ++r) {
    int row = 8 * half + r;
    int col = cb0 + (lane & 15);
    sH[row * 256 + col]      = (_Float16)fmaxf(c0[r] + b1[col], 0.0f);
    sH[row * 256 + col + 16] = (_Float16)fmaxf(c1[r] + b1[col + 16], 0.0f);
  }
  __syncthreads();

  // ---- GEMM2: 16x256 @ 256x128, each wave owns 16 msg columns
  v8f c = {};
  int cb = wave * 16;
#pragma unroll
  for (int kb = 0; kb < 256; kb += 32) {
    v16h a = frag_a_lds(sH, 256, 0, kb);
    v16h b = frag_b_swz(w2, 8, kb, cb);
    c = WMMA_F16(a, b, c);
  }
#pragma unroll
  for (int r = 0; r < 8; ++r) {
    int row = 8 * half + r;
    int col = cb + (lane & 15);
    float v = (c[r] + b2[col]) * sM[row];
    atomicAdd(&agg[(size_t)sTo[row] * D_ + col], v);
  }
}

// ---------------------------------------------------------------------------
// Update MLP (fused, residual). Block = 16 nodes, 256 threads.
//   A = [agg | h] (16 x 256); h += relu(A@W1+b1)@W2 + b2; refresh h16.
// ---------------------------------------------------------------------------
__global__ __launch_bounds__(256) void upd_kernel(
    float* __restrict__ h, _Float16* __restrict__ h16, const float* __restrict__ agg,
    const _Float16* __restrict__ w1, const float* __restrict__ b1,
    const _Float16* __restrict__ w2, const float* __restrict__ b2) {
  __shared__ __align__(16) _Float16 sA[16 * 256];
  __shared__ __align__(16) _Float16 sH[16 * 256];
  int tid = threadIdx.x;
  int n0  = blockIdx.x * 16;

  if (tid == 0 && n0 + 16 < NN_)
    __builtin_prefetch(&agg[(size_t)(n0 + 16) * D_], 0, 0);

  // agg half needs f32->f16 conversion; h half is a straight f16 copy.
  for (int t = tid; t < 16 * 128; t += 256) {
    int row = t >> 7, col = t & 127;
    sA[row * 256 + col] = (_Float16)agg[(size_t)(n0 + row) * D_ + col];
  }
  for (int t = tid; t < 16 * 16; t += 256) {
    int row = t >> 4, part = t & 15;
    ((uint4*)sA)[row * 32 + 16 + part] =
        ((const uint4*)(h16 + (size_t)(n0 + row) * D_))[part];
  }
  __syncthreads();

  int wave = tid >> 5, lane = tid & 31, half = lane >> 4;

  v8f c0 = {}, c1 = {};
  int cb0 = wave * 32;
#pragma unroll
  for (int kb = 0; kb < 256; kb += 32) {
    v16h a  = frag_a_lds(sA, 256, 0, kb);
    v16h b0 = frag_b_swz(w1, 16, kb, cb0);
    v16h b1v = frag_b_swz(w1, 16, kb, cb0 + 16);
    c0 = WMMA_F16(a, b0, c0);
    c1 = WMMA_F16(a, b1v, c1);
  }
#pragma unroll
  for (int r = 0; r < 8; ++r) {
    int row = 8 * half + r;
    int col = cb0 + (lane & 15);
    sH[row * 256 + col]      = (_Float16)fmaxf(c0[r] + b1[col], 0.0f);
    sH[row * 256 + col + 16] = (_Float16)fmaxf(c1[r] + b1[col + 16], 0.0f);
  }
  __syncthreads();

  v8f c = {};
  int cb = wave * 16;
#pragma unroll
  for (int kb = 0; kb < 256; kb += 32) {
    v16h a = frag_a_lds(sH, 256, 0, kb);
    v16h b = frag_b_swz(w2, 8, kb, cb);
    c = WMMA_F16(a, b, c);
  }
#pragma unroll
  for (int r = 0; r < 8; ++r) {
    int row = 8 * half + r;
    int col = cb + (lane & 15);
    size_t idx = (size_t)(n0 + row) * D_ + col;
    float v = h[idx] + c[r] + b2[col];
    h[idx] = v;
    h16[idx] = (_Float16)v;
  }
}

// ---------------------------------------------------------------------------
// transform(): relu(x@ft1+b1)@ft2+b2 for q/c embeddings -> tq16/tc16 (f16).
// Block = one (batch, q-or-c) pair; 30 rows padded to 32; 256 threads.
// ---------------------------------------------------------------------------
__global__ __launch_bounds__(256) void transform_kernel(
    const _Float16* __restrict__ h16,
    const _Float16* __restrict__ ft1, const float* __restrict__ b1,
    const _Float16* __restrict__ ft2, const float* __restrict__ b2,
    _Float16* __restrict__ tq16, _Float16* __restrict__ tc16) {
  __shared__ __align__(16) _Float16 sX[32 * 128];
  __shared__ __align__(16) _Float16 sH[32 * 64];
  int tid = threadIdx.x;
  int b   = blockIdx.x >> 1;
  int isC = blockIdx.x & 1;
  int base = b * NPG_ + isC * NC_;

  for (int t = tid; t < 32 * 16; t += 256) {
    int row = t >> 4, part = t & 15;
    uint4 v = {0u, 0u, 0u, 0u};
    if (row < NC_) v = ((const uint4*)(h16 + (size_t)(base + row) * D_))[part];
    ((uint4*)sX)[t] = v;
  }
  __syncthreads();

  int wave = tid >> 5, lane = tid & 31, half = lane >> 4;
  int rt = wave >> 2, ct = wave & 3;

  v8f c = {};
#pragma unroll
  for (int kb = 0; kb < 128; kb += 32) {
    v16h a = frag_a_lds(sX, 128, rt * 16, kb);
    v16h b0 = frag_b_swz(ft1, 4, kb, ct * 16);
    c = WMMA_F16(a, b0, c);
  }
#pragma unroll
  for (int r = 0; r < 8; ++r) {
    int row = rt * 16 + 8 * half + r;
    int col = ct * 16 + (lane & 15);
    sH[row * 64 + col] = (_Float16)fmaxf(c[r] + b1[col], 0.0f);
  }
  __syncthreads();

  v8f d = {};
#pragma unroll
  for (int kb = 0; kb < 64; kb += 32) {
    v16h a = frag_a_lds(sH, 64, rt * 16, kb);
    v16h b0 = frag_b_swz(ft2, 4, kb, ct * 16);
    d = WMMA_F16(a, b0, d);
  }
  _Float16* outp = isC ? tc16 : tq16;
#pragma unroll
  for (int r = 0; r < 8; ++r) {
    int row = rt * 16 + 8 * half + r;
    int col = ct * 16 + (lane & 15);
    float v = d[r] + b2[col];
    if (row >= NC_) v = 0.0f;
    if (!isC && row >= NQ_) v = 0.0f;   // q mask
    outp[((size_t)b * 32 + row) * T_ + col] = (_Float16)v;
  }
}

// ---------------------------------------------------------------------------
// Per-batch: log_alpha = tq@tc^T/0.1 via WMMA, 10 Sinkhorn iterations in LDS,
// score = -sum relu(q_emb - tp@c_emb). 128 threads (4 waves, 2x2 tiles).
// ---------------------------------------------------------------------------
__global__ __launch_bounds__(128) void sinkhorn_kernel(
    const _Float16* __restrict__ tq16, const _Float16* __restrict__ tc16,
    const float* __restrict__ h, float* __restrict__ out) {
  __shared__ __align__(16) _Float16 sQ[32 * 64];
  __shared__ __align__(16) _Float16 sC[32 * 64];
  __shared__ float sLA[32 * 32];
  __shared__ float sRed[128];
  int tid = threadIdx.x;
  int b   = blockIdx.x;

  for (int t = tid; t < 256; t += 128) {
    ((uint4*)sQ)[t] = ((const uint4*)(tq16 + (size_t)b * 32 * 64))[t];
    ((uint4*)sC)[t] = ((const uint4*)(tc16 + (size_t)b * 32 * 64))[t];
  }
  __syncthreads();

  int wave = tid >> 5, lane = tid & 31, half = lane >> 4;
  int rt = wave >> 1, ct = wave & 1;

  v8f c = {};
#pragma unroll
  for (int kb = 0; kb < 64; kb += 32) {
    v16h a = frag_a_lds(sQ, 64, rt * 16, kb);
    v16h bb = frag_bt_lds(sC, 64, kb, ct * 16);   // tc^T as B
    c = WMMA_F16(a, bb, c);
  }
#pragma unroll
  for (int r = 0; r < 8; ++r) {
    int row = rt * 16 + 8 * half + r;
    int col = ct * 16 + (lane & 15);
    sLA[row * 32 + col] = c[r] * 10.0f;   // 1 / SK_TEMP
  }
  __syncthreads();

  for (int it = 0; it < SKIT_; ++it) {
    if (tid < NC_) {            // normalize over c (axis 2)
      int q = tid;
      float mx = -1e30f;
      for (int cc = 0; cc < NC_; ++cc) mx = fmaxf(mx, sLA[q * 32 + cc]);
      float s = 0.0f;
      for (int cc = 0; cc < NC_; ++cc) s += __expf(sLA[q * 32 + cc] - mx);
      float lse = mx + __logf(s);
      for (int cc = 0; cc < NC_; ++cc) sLA[q * 32 + cc] -= lse;
    }
    __syncthreads();
    if (tid < NC_) {            // normalize over q (axis 1)
      int cc = tid;
      float mx = -1e30f;
      for (int q = 0; q < NC_; ++q) mx = fmaxf(mx, sLA[q * 32 + cc]);
      float s = 0.0f;
      for (int q = 0; q < NC_; ++q) s += __expf(sLA[q * 32 + cc] - mx);
      float lse = mx + __logf(s);
      for (int q = 0; q < NC_; ++q) sLA[q * 32 + cc] -= lse;
    }
    __syncthreads();
  }

  const float* qe = h + (size_t)(b * NPG_) * D_;
  const float* ce = h + (size_t)(b * NPG_ + NC_) * D_;
  float acc = 0.0f;
  for (int t = tid; t < NC_ * D_; t += 128) {
    int q = t >> 7, dd = t & 127;
    float mv = 0.0f;
    for (int cc = 0; cc < NC_; ++cc)
      mv += __expf(sLA[q * 32 + cc]) * ce[(size_t)cc * D_ + dd];
    acc += fmaxf(qe[(size_t)q * D_ + dd] - mv, 0.0f);
  }
  sRed[tid] = acc;
  __syncthreads();
  for (int s = 64; s > 0; s >>= 1) {
    if (tid < s) sRed[tid] += sRed[tid + s];
    __syncthreads();
  }
  if (tid == 0) out[b] = -sRed[0];
}

// ---------------------------------------------------------------------------
extern "C" void kernel_launch(void* const* d_in, const int* in_sizes, int n_in,
                              void* d_out, int out_size, void* d_ws, size_t ws_size,
                              hipStream_t stream) {
  const float* node_f   = (const float*)d_in[0];
  const float* edge_f   = (const float*)d_in[1];
  const float* mask     = (const float*)d_in[2];
  const float* enc_nw   = (const float*)d_in[3];
  const float* enc_nb   = (const float*)d_in[4];
  const float* enc_ew   = (const float*)d_in[5];
  const float* enc_eb   = (const float*)d_in[6];
  const float* msg_w1   = (const float*)d_in[7];
  const float* msg_b1   = (const float*)d_in[8];
  const float* msg_w2   = (const float*)d_in[9];
  const float* msg_b2   = (const float*)d_in[10];
  const float* upd_w1   = (const float*)d_in[11];
  const float* upd_b1   = (const float*)d_in[12];
  const float* upd_w2   = (const float*)d_in[13];
  const float* upd_b2   = (const float*)d_in[14];
  const float* ft1_w    = (const float*)d_in[15];
  const float* ft1_b    = (const float*)d_in[16];
  const float* ft2_w    = (const float*)d_in[17];
  const float* ft2_b    = (const float*)d_in[18];
  const int*   from_idx = (const int*)d_in[19];
  const int*   to_idx   = (const int*)d_in[20];
  float* scores = (float*)d_out;

  char* ws = (char*)d_ws;
  size_t off = 0;
  auto carve = [&](size_t bytes) {
    void* p = ws + off;
    off = (off + bytes + 255) & ~(size_t)255;
    return p;
  };
  float*    h    = (float*)carve((size_t)NN_ * D_ * 4);
  _Float16* h16  = (_Float16*)carve((size_t)NN_ * D_ * 2);
  _Float16* e16  = (_Float16*)carve((size_t)EE_ * D_ * 2);
  float*    agg  = (float*)carve((size_t)NN_ * D_ * 4);
  _Float16* tq16 = (_Float16*)carve((size_t)B_ * 32 * T_ * 2);
  _Float16* tc16 = (_Float16*)carve((size_t)B_ * 32 * T_ * 2);
  _Float16* w1s  = (_Float16*)carve((size_t)384 * 256 * 2);
  _Float16* w2s  = (_Float16*)carve((size_t)256 * 128 * 2);
  _Float16* u1s  = (_Float16*)carve((size_t)256 * 256 * 2);
  _Float16* u2s  = (_Float16*)carve((size_t)256 * 128 * 2);
  _Float16* f1s  = (_Float16*)carve((size_t)128 * 64 * 2);
  _Float16* f2s  = (_Float16*)carve((size_t)64 * 64 * 2);
  (void)ws_size; (void)n_in; (void)in_sizes; (void)out_size;

  auto swz = [&](const float* src, _Float16* dst, int K, int ncols) {
    int n = K * ncols;
    swizzle_w_kernel<<<(n + 255) / 256, 256, 0, stream>>>(src, dst, K, ncols);
  };
  swz(msg_w1, w1s, 384, 256);
  swz(msg_w2, w2s, 256, 128);
  swz(upd_w1, u1s, 256, 256);
  swz(upd_w2, u2s, 256, 128);
  swz(ft1_w,  f1s, 128, 64);
  swz(ft2_w,  f2s, 64, 64);

  encode_node_kernel<<<(NN_ * D_ + 255) / 256, 256, 0, stream>>>(node_f, enc_nw, enc_nb, h, h16, NN_);
  encode_edge_kernel<<<(EE_ * D_ + 255) / 256, 256, 0, stream>>>(edge_f, enc_ew, enc_eb, e16, EE_);

  for (int p = 0; p < NPROP_; ++p) {
    zero_kernel<<<(NN_ * D_ + 255) / 256, 256, 0, stream>>>(agg, NN_ * D_);
    msg_kernel<<<EE_ / 16, 256, 0, stream>>>(h16, e16, from_idx, to_idx, mask,
                                             w1s, msg_b1, w2s, msg_b2, agg);
    upd_kernel<<<NN_ / 16, 256, 0, stream>>>(h, h16, agg, u1s, upd_b1, u2s, upd_b2);
  }

  transform_kernel<<<B_ * 2, 256, 0, stream>>>(h16, f1s, ft1_b, f2s, ft2_b, tq16, tc16);
  sinkhorn_kernel<<<B_, 128, 0, stream>>>(tq16, tc16, h, scores);
}
